// MultiClassDiceLoss_59803124629506
// MI455X (gfx1250) — matrix operand order repaired and checked by reading, not compiled, not run on hardware
//
#include <hip/hip_runtime.h>
#include <hip/hip_bf16.h>

// MultiClassDiceLoss — HBM-bound counting reduction for MI455X (gfx1250, wave32).
// 12 global counters (per class: |x==c&valid|, |t==c&valid|, |x==c&t==c&valid|),
// accumulated as byte-packed per-lane counts, reduced across the wave with one
// v_wmma_f32_16x16x32_f16 (A = per-lane counters, B = 0/1 selection matrix).
// Loads are non-temporal (single-pass stream, no reuse) and issued 8-deep per
// unrolled iteration for memory-level parallelism.

typedef __attribute__((ext_vector_type(16))) _Float16 v16h;
typedef __attribute__((ext_vector_type(8)))  float    v8f;
typedef __attribute__((ext_vector_type(4)))  float    fvec4;   // clang ext-vector: OK for nontemporal builtin
typedef __attribute__((ext_vector_type(4)))  int      ivec4;

#define IGNORE_INDEX (-100)
#define BLOCK 256
#define UNROLL 4
#define CHUNKS_PER_THREAD 16   // 16 float4 chunks = 64 elems/thread -> counts <= 64 (exact in f16/byte)

__global__ __launch_bounds__(BLOCK)
void dice_count_kernel(const float* __restrict__ x, const int* __restrict__ t,
                       float* __restrict__ gacc, int nchunks, int n)
{
    const int tid      = blockIdx.x * BLOCK + threadIdx.x;
    const int nthreads = gridDim.x * BLOCK;

    // Byte-packed counters: byte c of accX = count(x==c & valid), etc.  (counts <= 64 < 256)
    unsigned accX = 0u, accT = 0u, accI = 0u;

    const fvec4* __restrict__ x4 = reinterpret_cast<const fvec4*>(x);
    const ivec4* __restrict__ t4 = reinterpret_cast<const ivec4*>(t);

#define PROC(XF, TI) {                                              \
        const int      ti    = (TI);                                \
        const bool     valid = (ti != IGNORE_INDEX);                \
        const int      xic   = (int)(XF);            /* 0..3 */     \
        const unsigned bx    = 1u << ((((unsigned)xic) & 3u) << 3); \
        const unsigned bt    = 1u << ((((unsigned)ti)  & 3u) << 3); \
        accX += valid ? bx : 0u;                                    \
        accT += valid ? bt : 0u;                                    \
        accI += (ti == xic) ? bx : 0u;  /* equality implies valid: -100 not in 0..3 */ \
    }

    int i = tid;
    // Main unrolled loop: issue all 8 non-temporal b128 loads before processing.
    for (; i + (UNROLL - 1) * nthreads < nchunks; i += UNROLL * nthreads) {
        fvec4 xv[UNROLL];
        ivec4 tv[UNROLL];
#pragma unroll
        for (int u = 0; u < UNROLL; ++u) {
            xv[u] = __builtin_nontemporal_load(&x4[i + u * nthreads]);
            tv[u] = __builtin_nontemporal_load(&t4[i + u * nthreads]);
        }
#pragma unroll
        for (int u = 0; u < UNROLL; ++u) {
            PROC(xv[u][0], tv[u][0]) PROC(xv[u][1], tv[u][1])
            PROC(xv[u][2], tv[u][2]) PROC(xv[u][3], tv[u][3])
        }
    }
    // Remainder chunks (empty when nchunks % (UNROLL*nthreads) == 0).
    for (; i < nchunks; i += nthreads) {
        fvec4 xv = __builtin_nontemporal_load(&x4[i]);
        ivec4 tv = __builtin_nontemporal_load(&t4[i]);
        PROC(xv[0], tv[0]) PROC(xv[1], tv[1]) PROC(xv[2], tv[2]) PROC(xv[3], tv[3])
    }

    // Scalar tail (N % 4 elements), handled by the first threads.
    {
        const int base = nchunks * 4;
        const int rem  = n - base;
        if (tid < rem) {
            const float xf = x[base + tid];
            PROC(xf, t[base + tid])
        }
    }
#undef PROC

    // ---- Wave-level reduce of the 12 counters via WMMA ---------------------
    // cnt[j]: j=0..3 -> X_c, 4..7 -> T_c, 8..11 -> I_c. All <= 64: exact in f16.
    // A fragment: lane puts cnt[j] in half j (halves 12..15 = 0).
    // B fragment: selection matrix so that D[m][n] = cnt_n(lane m) + cnt_n(lane m+16).
    //   B layout (32x16 f16): lanes 0-15 hold K=0..7,16..23 of column N=lane (half h <-> K=h, h>=8: K=h+8);
    //   lanes 16-31 hold K=8..15,24..31 of column N=lane-16.
    //   Counter j lives at A-columns K: {j, j+8} (j<8) and {j+8, j+16} (j>=8).
    v16h a;
#pragma unroll
    for (int j = 0; j < 16; ++j) {
        unsigned v = 0u;
        if (j < 4)       v = (accX >> (8 * j)) & 0xFFu;
        else if (j < 8)  v = (accT >> (8 * (j - 4))) & 0xFFu;
        else if (j < 12) v = (accI >> (8 * (j - 8))) & 0xFFu;
        a[j] = (_Float16)(float)v;
    }

    const int lane  = threadIdx.x & 31;
    const int ncol  = lane & 15;
    const int kbase = (lane & 16) ? 16 : 0;
    v16h b;
#pragma unroll
    for (int h = 0; h < 16; ++h) {
        const int K = (h < 8) ? (kbase >> 1) + h : 16 + (kbase >> 1) + (h - 8);
        int ctr;
        if (K < 16)                 ctr = K & 7;        // K 0..7 -> K ; K 8..15 -> K-8
        else if (K < 20)            ctr = K - 8;        // 16..19 -> 8..11
        else if (K >= 24 && K < 28) ctr = K - 16;       // 24..27 -> 8..11
        else                        ctr = -1;           // unused slots
        b[h] = (ctr == ncol) ? (_Float16)1.0f : (_Float16)0.0f;
    }

    v8f acc = {};
    // D = A x B : exact (small-integer f16 inputs, f32 accumulate). EXEC is all-ones here.
    acc = __builtin_amdgcn_wmma_f32_16x16x32_f16(false, a, false, b,
                                                 (short)0, acc, false, false);
    float s = acc[0] + acc[1] + acc[2] + acc[3] + acc[4] + acc[5] + acc[6] + acc[7];
    // s(lane) holds counter (lane&15) summed over half the rows; add the partner half.
    s += __shfl_xor(s, 16, 32);
    // Now lanes 0..11 hold the full 32-lane totals for counters 0..11.

    __shared__ float red[12];
    if (threadIdx.x < 12) red[threadIdx.x] = 0.0f;
    __syncthreads();
    if (lane < 12) atomicAdd(&red[lane], s);            // ds_add_f32
    __syncthreads();
    if (threadIdx.x < 12) atomicAdd(&gacc[threadIdx.x], red[threadIdx.x]);  // global_atomic_add_f32
}

__global__ void dice_finalize_kernel(const float* __restrict__ gacc, float* __restrict__ out)
{
    if (blockIdx.x == 0 && threadIdx.x == 0) {
        float lsum = 0.0f;
#pragma unroll
        for (int c = 0; c < 4; ++c) {
            const float X = gacc[c];
            const float T = gacc[4 + c];
            const float I = gacc[8 + c];
            const float dice = (2.0f * I + 1.0f) / (X + T + 1.0f);
            lsum += 1.0f - dice;
        }
        out[0] = 0.25f * lsum;
    }
}

extern "C" void kernel_launch(void* const* d_in, const int* in_sizes, int n_in,
                              void* d_out, int out_size, void* d_ws, size_t ws_size,
                              hipStream_t stream)
{
    const float* x   = (const float*)d_in[0];
    const int*   t   = (const int*)d_in[1];
    float*       acc = (float*)d_ws;
    const int    n   = in_sizes[0];

    // Zero the 12 global accumulators each call (graph-capture safe).
    (void)hipMemsetAsync(d_ws, 0, 12 * sizeof(float), stream);

    const int nchunks = n / 4;
    int blocks = (nchunks + BLOCK * CHUNKS_PER_THREAD - 1) / (BLOCK * CHUNKS_PER_THREAD);
    if (blocks < 1) blocks = 1;   // n = 16.7M -> 1024 blocks, 8192 waves, <=64 elems/thread

    dice_count_kernel<<<blocks, BLOCK, 0, stream>>>(x, t, acc, nchunks, n);
    dice_finalize_kernel<<<1, 32, 0, stream>>>(acc, (float*)d_out);
}
